// PixelLSTM_61838939128410
// MI455X (gfx1250) — matrix operand order repaired and checked by reading, not compiled, not run on hardware
//
#include <hip/hip_runtime.h>

typedef __attribute__((ext_vector_type(16))) __bf16 v16bf;
typedef __attribute__((ext_vector_type(8)))  __bf16 v8bf;
typedef __attribute__((ext_vector_type(8)))  float  v8f;

// Explicit global-address-space pointer for the swizzled weights: keeps the
// loads as global_load_b128 (LOADcnt only) even after pointer laundering.
typedef const __attribute__((address_space(1))) __bf16 gc_bf16;
typedef gc_bf16* gbfp;
typedef const __attribute__((address_space(1))) v8bf gc_v8bf;

#define Bq 512
#define Tq 365
#define Dq 16
#define Hq 256
#define HSTR 264   // LDS row stride (bf16 elems) for h tiles, padded vs 256 for banking

// ---------------------------------------------------------------------------
// Weight swizzle: W [1024 x Ksrc] (row n = gate output, col k) -> B fragments.
// Fragment (nt, kt): 32 lanes x 16 bf16. Lane l holds B[k = kt*32 + (l>>4)*8 + (j&7) + (j>>3)*16,
// n = nt*16 + (l&15)] for j = 0..15 (2 bf16 per VGPR, j ascending) per CDNA5 16-bit B layout.
// frag index = nt*KT + kt, 512 bf16 (1KB) each. k >= Ksrc zero-padded.
// ---------------------------------------------------------------------------
__global__ void swizzleW_kernel(const float* __restrict__ W, __bf16* __restrict__ dst,
                                int KT, int Ksrc) {
    int i = blockIdx.x * blockDim.x + threadIdx.x;
    int total = 64 * KT * 512;
    if (i >= total) return;
    int j  = i & 15;
    int l  = (i >> 4) & 31;
    int f  = i >> 9;
    int kt = f % KT;
    int nt = f / KT;
    int n = nt * 16 + (l & 15);
    int k = kt * 32 + (l >> 4) * 8 + (j & 7) + (j >> 3) * 16;
    float v = (k < Ksrc) ? W[(size_t)n * Ksrc + k] : 0.0f;
    dst[i] = (__bf16)v;
}

// h0 = c0 = x_static @ Ws^T + bs  -> f32 (for c) and bf16 (for h)
__global__ void init_hc_kernel(const float* __restrict__ xs, const float* __restrict__ Ws,
                               const float* __restrict__ bs,
                               float* __restrict__ cinit, __bf16* __restrict__ hinit) {
    int i = blockIdx.x * blockDim.x + threadIdx.x;
    if (i >= Bq * Hq) return;
    int b = i >> 8, j = i & 255;
    float s = bs[j];
    #pragma unroll
    for (int k = 0; k < 32; k++) s += xs[b * 32 + k] * Ws[j * 32 + k];
    cinit[i] = s;
    hinit[i] = (__bf16)s;
}

// ---------------------------------------------------------------------------
// Fragment loaders
// ---------------------------------------------------------------------------
// A (16x32 MxK, 16-bit): lane<16 -> M=lane, K = kh*8 + {0..7} (elems 0-7) and
// 16 + kh*8 + {0..7} (elems 8-15), kh = lane>>4. Two 16B LDS loads per lane.
__device__ __forceinline__ v16bf loadA(const __bf16* base, int stride, int kt, int lane) {
    int m = lane & 15, kh = lane >> 4;
    const __bf16* p = base + m * stride + kt * 32 + kh * 8;
    v8bf lo = *(const v8bf*)p;
    v8bf hi = *(const v8bf*)(p + 16);
    v16bf a;
    #pragma unroll
    for (int e = 0; e < 8; e++) { a[e] = lo[e]; a[e + 8] = hi[e]; }
    return a;
}

// B fragment: 32 contiguous bytes per lane from the swizzled weight buffer (AS1).
__device__ __forceinline__ v16bf loadB(gbfp frag, int lane) {
    gc_v8bf* p = (gc_v8bf*)(frag + lane * 16);
    v8bf lo = p[0];
    v8bf hi = p[1];
    v16bf b;
    #pragma unroll
    for (int e = 0; e < 8; e++) { b[e] = lo[e]; b[e + 8] = hi[e]; }
    return b;
}

__device__ __forceinline__ float sigf(float x)  { return 1.0f / (1.0f + __expf(-x)); }
__device__ __forceinline__ float tanhfast(float x) {
    float e = __expf(2.0f * x);          // robust at +/-inf
    return 1.0f - 2.0f / (e + 1.0f);
}

// ---------------------------------------------------------------------------
// One LSTM layer step for a 16-row batch tile.
// Wave w owns h columns [w*64, w*64+64) as 4 n-groups; the 4 gate tiles
// (i,f,g,o) of a group live in one wave so the c/h update is register-local.
// Unified K-step sequence: s in [0,KTIN) = input path, [KTIN,KTIN+8) =
// recurrent path. Ping-pong (parity-indexed) operand buffers: fragments for
// step s+1 issue before the WMMAs of step s; constant indices after full
// unroll mean no register copies. Accumulators start at 0 (inline SRC2=0 on
// the first WMMA); bias is folded in during the gate phase from registers.
// ---------------------------------------------------------------------------
template<int KTIN, bool DO_OUT>
__device__ __forceinline__ void lstm_layer_step(
    const __bf16* Ain, int aStride,
    gbfp wIn,
    const __bf16* hPrev, gbfp wRec,
    const float (&bR)[4][4], float (&c)[4][8],
    __bf16* hNext,
    const float (&wor)[4], float* outw,
    int w, int lane)
{
    constexpr int NSTEP = KTIN + 8;
    float rowp[8];
    #pragma unroll
    for (int v = 0; v < 8; v++) rowp[v] = 0.0f;

    #pragma unroll
    for (int ng = 0; ng < 4; ng++) {
        const int hb   = w * 4 + ng;           // absolute h-column tile 0..15
        const int ncol = hb * 16 + (lane & 15);

        auto ldA = [&](int s) -> v16bf {
            return (s < KTIN) ? loadA(Ain, aStride, s, lane)
                              : loadA(hPrev, HSTR, s - KTIN, lane);
        };
        auto ldB = [&](int s, int g) -> v16bf {
            gbfp base = (s < KTIN)
                ? wIn  + ((size_t)((g * 16 + hb) * KTIN + s) << 9)
                : wRec + ((size_t)((g * 16 + hb) * 8 + (s - KTIN)) << 9);
            return loadB(base, lane);
        };

        v8f acc[4];
        #pragma unroll
        for (int g = 0; g < 4; g++)
            #pragma unroll
            for (int e = 0; e < 8; e++) acc[g][e] = 0.0f;

        v16bf abuf[2];
        v16bf bbuf[2][4];
        abuf[0] = ldA(0);
        #pragma unroll
        for (int g = 0; g < 4; g++) bbuf[0][g] = ldB(0, g);

        #pragma unroll
        for (int s = 0; s < NSTEP; s++) {
            const int cb = s & 1, nb = cb ^ 1;
            if (s + 1 < NSTEP) {                 // prefetch next K-step
                abuf[nb] = ldA(s + 1);
                #pragma unroll
                for (int g = 0; g < 4; g++) bbuf[nb][g] = ldB(s + 1, g);
            }
            #pragma unroll
            for (int g = 0; g < 4; g++)
                acc[g] = __builtin_amdgcn_wmma_f32_16x16x32_bf16(
                    false, abuf[cb], false, bbuf[cb][g], (short)0, acc[g], false, false);
        }

        // gates + state update; D layout: elem (m,n): m = v + 8*(lane>>4), n = ncol
        #pragma unroll
        for (int v = 0; v < 8; v++) {
            float ig = sigf(acc[0][v] + bR[ng][0]);
            float fg = sigf(acc[1][v] + bR[ng][1]);
            float gg = tanhfast(acc[2][v] + bR[ng][2]);
            float og = sigf(acc[3][v] + bR[ng][3]);
            float cc = fg * c[ng][v] + ig * gg;
            c[ng][v] = cc;
            float hv = og * tanhfast(cc);
            int m = v + ((lane >> 4) << 3);
            hNext[m * HSTR + ncol] = (__bf16)hv;
            if (DO_OUT) {
                float p = hv * wor[ng];
                p += __shfl_xor(p, 1);
                p += __shfl_xor(p, 2);
                p += __shfl_xor(p, 4);
                p += __shfl_xor(p, 8);     // full sum within each 16-lane half
                rowp[v] += p;
            }
        }
    }
    if (DO_OUT && ((lane & 15) == 0)) {
        int mbase = (lane >> 4) << 3;
        #pragma unroll
        for (int v = 0; v < 8; v++) outw[w * 16 + mbase + v] = rowp[v];
    }
}

// ---------------------------------------------------------------------------
// Fused 2-layer LSTM + output projection. 32 blocks x 16 batch rows.
// ---------------------------------------------------------------------------
__global__ __launch_bounds__(128) void lstm_fused_kernel(
    const float* __restrict__ x,
    const __bf16* __restrict__ w0in, const __bf16* __restrict__ w0rec,
    const __bf16* __restrict__ w1in, const __bf16* __restrict__ w1rec,
    const float* __restrict__ bih0, const float* __restrict__ bhh0,
    const float* __restrict__ bih1, const float* __restrict__ bhh1,
    const float* __restrict__ Wo, const float* __restrict__ bo,
    const float* __restrict__ cinit, const __bf16* __restrict__ hinit,
    float* __restrict__ out)
{
    __shared__ __bf16 xbuf[16 * 32];            // K padded 16 -> 32, upper half zero
    __shared__ __bf16 h0buf[2][16 * HSTR];
    __shared__ __bf16 h1buf[2][16 * HSTR];
    __shared__ float  outw[4 * 16];             // per-wave output partials

    const int tid  = threadIdx.x;
    const int lane = tid & 31;
    const int w    = tid >> 5;
    const int wb   = blockIdx.x * 16;           // batch base row

    for (int i = tid; i < 16 * 32; i += 128) xbuf[i] = (__bf16)0.0f;
    for (int i = tid; i < 16 * 256; i += 128) {
        int r = i >> 8, cix = i & 255;
        __bf16 hv = hinit[(size_t)(wb + r) * Hq + cix];
        h0buf[0][r * HSTR + cix] = hv;
        h1buf[0][r * HSTR + cix] = hv;
    }

    // per-wave register-resident constants: biases, Wo slice, c state
    float b0r[4][4], b1r[4][4], wor[4];
    float c0[4][8], c1[4][8];
    #pragma unroll
    for (int ng = 0; ng < 4; ng++) {
        int ncol = (w * 4 + ng) * 16 + (lane & 15);
        #pragma unroll
        for (int g = 0; g < 4; g++) {
            b0r[ng][g] = bih0[g * 256 + ncol] + bhh0[g * 256 + ncol];
            b1r[ng][g] = bih1[g * 256 + ncol] + bhh1[g * 256 + ncol];
        }
        wor[ng] = Wo[ncol];
        #pragma unroll
        for (int v = 0; v < 8; v++) {
            int m = v + ((lane >> 4) << 3);
            float cv = cinit[(size_t)(wb + m) * Hq + ncol];
            c0[ng][v] = cv;
            c1[ng][v] = cv;
        }
    }
    float boV = bo[0];
    __syncthreads();

    // AS1-typed weight pointers, laundered every iteration: loads stay
    // global_load_b128 and cannot be hoisted out of the time loop (keeps
    // fragments streaming from L2 instead of being spilled to scratch).
    gbfp w0in_t  = (gbfp)w0in;
    gbfp w0rec_t = (gbfp)w0rec;
    gbfp w1in_t  = (gbfp)w1in;
    gbfp w1rec_t = (gbfp)w1rec;

    int cur = 0;
    for (int t = 0; t < Tq; t++) {
        asm volatile("" : "+s"(w0in_t), "+s"(w0rec_t), "+s"(w1in_t), "+s"(w1rec_t));

        // stage x_t tile (16 rows x 16 cols) as bf16
        for (int e = tid; e < 256; e += 128) {
            int r = e >> 4, d = e & 15;
            xbuf[r * 32 + d] = (__bf16)x[(size_t)(wb + r) * (Tq * Dq) + (size_t)t * Dq + d];
        }
        __syncthreads();

        lstm_layer_step<1, false>(xbuf, 32, w0in_t,
                                  h0buf[cur], w0rec_t, b0r, c0, h0buf[cur ^ 1],
                                  wor, nullptr, w, lane);
        __syncthreads();

        lstm_layer_step<8, true>(h0buf[cur ^ 1], HSTR, w1in_t,
                                 h1buf[cur], w1rec_t, b1r, c1, h1buf[cur ^ 1],
                                 wor, outw, w, lane);
        __syncthreads();

        if (tid < 16) {
            float s = outw[tid] + outw[16 + tid] + outw[32 + tid] + outw[48 + tid];
            out[(size_t)(wb + tid) * Tq + t] = s + boV;
        }
        cur ^= 1;
    }
}

// ---------------------------------------------------------------------------
extern "C" void kernel_launch(void* const* d_in, const int* in_sizes, int n_in,
                              void* d_out, int out_size, void* d_ws, size_t ws_size,
                              hipStream_t stream) {
    const float* x    = (const float*)d_in[0];
    const float* xs   = (const float*)d_in[1];
    const float* Wih0 = (const float*)d_in[2];
    const float* Whh0 = (const float*)d_in[3];
    const float* bih0 = (const float*)d_in[4];
    const float* bhh0 = (const float*)d_in[5];
    const float* Wih1 = (const float*)d_in[6];
    const float* Whh1 = (const float*)d_in[7];
    const float* bih1 = (const float*)d_in[8];
    const float* bhh1 = (const float*)d_in[9];
    const float* Ws   = (const float*)d_in[10];
    const float* bs   = (const float*)d_in[11];
    const float* Wo   = (const float*)d_in[12];
    const float* bo   = (const float*)d_in[13];
    float* out = (float*)d_out;

    char* ws = (char*)d_ws;
    const size_t SZ_W0IN  = (size_t)64  * 512 * 2;   //  64 KB
    const size_t SZ_WREC  = (size_t)512 * 512 * 2;   // 512 KB each
    __bf16* w0in  = (__bf16*)(ws);
    __bf16* w0rec = (__bf16*)(ws + SZ_W0IN);
    __bf16* w1in  = (__bf16*)(ws + SZ_W0IN + SZ_WREC);
    __bf16* w1rec = (__bf16*)(ws + SZ_W0IN + 2 * SZ_WREC);
    float*  cinit = (float*) (ws + SZ_W0IN + 3 * SZ_WREC);
    __bf16* hinit = (__bf16*)(ws + SZ_W0IN + 3 * SZ_WREC + (size_t)Bq * Hq * 4);

    // weight swizzle (weights live in L2 afterwards; main loop streams them)
    swizzleW_kernel<<<(64 * 1 * 512 + 255) / 256, 256, 0, stream>>>(Wih0, w0in, 1, Dq);
    swizzleW_kernel<<<(64 * 8 * 512 + 255) / 256, 256, 0, stream>>>(Whh0, w0rec, 8, Hq);
    swizzleW_kernel<<<(64 * 8 * 512 + 255) / 256, 256, 0, stream>>>(Wih1, w1in, 8, Hq);
    swizzleW_kernel<<<(64 * 8 * 512 + 255) / 256, 256, 0, stream>>>(Whh1, w1rec, 8, Hq);
    init_hc_kernel<<<(Bq * Hq + 255) / 256, 256, 0, stream>>>(xs, Ws, bs, cinit, hinit);

    lstm_fused_kernel<<<Bq / 16, 128, 0, stream>>>(
        x, w0in, w0rec, w1in, w1rec,
        bih0, bhh0, bih1, bhh1, Wo, bo, cinit, hinit, out);
}
